// MAPDP_23021024706729
// MI455X (gfx1250) — matrix operand order repaired and checked by reading, not compiled, not run on hardware
//
#include <hip/hip_runtime.h>
#include <hip/hip_bf16.h>
#include <math.h>

#define B_  16
#define R_  384
#define NN_ 1024
#define S_  32
#define E_  256
#define H_  8
#define L_  3
#define RD_ 16
#define HD_ 32
#define BR_ (B_ * R_)

typedef _Float16 h16;
typedef __attribute__((ext_vector_type(16))) _Float16 v16h;
typedef __attribute__((ext_vector_type(8)))  float    v8f;

#define WMMA_F16(a, b, c) \
  __builtin_amdgcn_wmma_f32_16x16x32_f16(false, (a), false, (b), (short)0, (c), false, false)

// ---- async global->LDS staging (gfx1250 ASYNCcnt path), guarded ------------
#if defined(__has_builtin)
#  if __has_builtin(__builtin_amdgcn_global_load_async_to_lds_b128) && \
      __has_builtin(__builtin_amdgcn_s_wait_asynccnt)
#    define HAVE_ASYNC_LDS 1
#  endif
#endif

typedef int v4i_ __attribute__((vector_size(4 * sizeof(int))));
typedef __attribute__((address_space(1))) v4i_ as1_v4i;
typedef __attribute__((address_space(3))) v4i_ as3_v4i;

__device__ __forceinline__ void lds_copy16(void* lp, const void* gp) {
#ifdef HAVE_ASYNC_LDS
  __builtin_amdgcn_global_load_async_to_lds_b128(
      (as1_v4i*)(uintptr_t)gp,
      (as3_v4i*)(uint32_t)(uintptr_t)lp,
      0, 0);
#else
  *reinterpret_cast<int4*>(lp) = *reinterpret_cast<const int4*>(gp);
#endif
}
__device__ __forceinline__ void lds_copy_fence() {
#ifdef HAVE_ASYNC_LDS
  __builtin_amdgcn_s_wait_asynccnt(0);
#endif
  __syncthreads();
}

// ---------------- WMMA fragment loaders (contiguous k = 16*hi + s, wave32) -------

// A: 16x32 f16, row-major, lda = row stride. lane m = l&15, hi = l>>4.
__device__ __forceinline__ v16h ld_a(const h16* __restrict__ A, int lda, int lane) {
  int m = lane & 15, hi = (lane >> 4) & 1;
  const h16* row = A + (size_t)m * lda;
  v16h a;
#pragma unroll
  for (int i = 0; i < 4; ++i) {
    int k0 = hi * 16 + 2 * i;         // VGPR 0..3: K = 16*hi + 2i, +1
    a[2 * i]     = row[k0];
    a[2 * i + 1] = row[k0 + 1];
    int k1 = hi * 16 + 8 + 2 * i;     // VGPR 4..7: K = 16*hi + 8 + 2i, +1
    a[8 + 2 * i]     = row[k1];
    a[8 + 2 * i + 1] = row[k1 + 1];
  }
  return a;
}

// B: 32x16 f16 where element(k,n) = Bt[n*ldb + k]  (K-pairs contiguous).
__device__ __forceinline__ v16h ld_bt(const h16* __restrict__ Bt, int ldb, int lane) {
  int n = lane & 15, hi = (lane >> 4) & 1;
  const h16* col = Bt + (size_t)n * ldb;
  v16h b;
#pragma unroll
  for (int j = 0; j < 8; ++j) {
    int k0 = hi * 16 + 2 * j;         // VGPR j: K = 16*hi + 2j, +1
    b[2 * j]     = col[k0];
    b[2 * j + 1] = col[k0 + 1];
  }
  return b;
}

// B: 32x16 f16 row-major: element(k,n) = Bm[k*ldb + n].
__device__ __forceinline__ v16h ld_b(const h16* __restrict__ Bm, int ldb, int lane) {
  int n = lane & 15, hi = (lane >> 4) & 1;
  v16h b;
#pragma unroll
  for (int j = 0; j < 8; ++j) {
    int k0 = hi * 16 + 2 * j;
    b[2 * j]     = Bm[(size_t)k0 * ldb + n];
    b[2 * j + 1] = Bm[(size_t)(k0 + 1) * ldb + n];
  }
  return b;
}

__device__ __forceinline__ float wave_sum(float s) {
#pragma unroll
  for (int off = 16; off; off >>= 1) s += __shfl_xor(s, off, 32);
  return s;
}
__device__ __forceinline__ float wave_max(float s) {
#pragma unroll
  for (int off = 16; off; off >>= 1) s = fmaxf(s, __shfl_xor(s, off, 32));
  return s;
}

// ---------------- Kernels ---------------------------------------------------

// h0 = concat(fc,tc) @ Wreq + breq ; also stash fc/tc
__global__ void embed_kernel(const float* __restrict__ coords,
                             const int* __restrict__ from_idx, const int* __restrict__ to_idx,
                             const float* __restrict__ Wreq, const float* __restrict__ breq,
                             float* __restrict__ h, h16* __restrict__ hh,
                             float* __restrict__ fc, float* __restrict__ tc) {
  int idx = blockIdx.x * blockDim.x + threadIdx.x;
  if (idx >= BR_ * E_) return;
  int e = idx & (E_ - 1);
  int br = idx / E_;
  int b  = br / R_;
  int f = from_idx[br], t = to_idx[br];
  float fx = coords[((size_t)b * NN_ + f) * 2 + 0];
  float fy = coords[((size_t)b * NN_ + f) * 2 + 1];
  float tx = coords[((size_t)b * NN_ + t) * 2 + 0];
  float ty = coords[((size_t)b * NN_ + t) * 2 + 1];
  float x = fx * Wreq[0 * E_ + e] + fy * Wreq[1 * E_ + e] +
            tx * Wreq[2 * E_ + e] + ty * Wreq[3 * E_ + e] + breq[e];
  h[idx]  = x;
  hh[idx] = (h16)x;
  if (e == 0) {
    fc[br * 2 + 0] = fx; fc[br * 2 + 1] = fy;
    tc[br * 2 + 0] = tx; tc[br * 2 + 1] = ty;
  }
}

// qkfeat[b,r,s,:] = relu(d * relW1 + relb1) @ relW2 + relb2   (E-dim fused)
__global__ void qkfeat_kernel(const float* __restrict__ fc,
                              const float* __restrict__ relW1, const float* __restrict__ relb1,
                              const float* __restrict__ relW2, const float* __restrict__ relb2,
                              h16* __restrict__ qkfeat) {
  __shared__ __align__(16) float w1[E_], b1[E_], w2[E_ * RD_], b2s[RD_];
  {
    int t = threadIdx.x;
    for (int i = t; i < E_ / 4; i += blockDim.x)       lds_copy16(&w1[i * 4],  &relW1[i * 4]);
    for (int i = t; i < E_ / 4; i += blockDim.x)       lds_copy16(&b1[i * 4],  &relb1[i * 4]);
    for (int i = t; i < E_ * RD_ / 4; i += blockDim.x) lds_copy16(&w2[i * 4],  &relW2[i * 4]);
    for (int i = t; i < RD_ / 4; i += blockDim.x)      lds_copy16(&b2s[i * 4], &relb2[i * 4]);
  }
  lds_copy_fence();
  size_t idx = (size_t)blockIdx.x * blockDim.x + threadIdx.x;
  if (idx >= (size_t)B_ * R_ * R_) return;
  int s = (int)(idx % R_);
  size_t t = idx / R_;
  int r = (int)(t % R_);
  int b = (int)(t / R_);
  float dx = fc[((size_t)b * R_ + r) * 2 + 0] - fc[((size_t)b * R_ + s) * 2 + 0];
  float dy = fc[((size_t)b * R_ + r) * 2 + 1] - fc[((size_t)b * R_ + s) * 2 + 1];
  float d = sqrtf(dx * dx + dy * dy);
  float acc[RD_];
#pragma unroll
  for (int f = 0; f < RD_; ++f) acc[f] = b2s[f];
  for (int e = 0; e < E_; ++e) {
    float a = fmaxf(d * w1[e] + b1[e], 0.f);
#pragma unroll
    for (int f = 0; f < RD_; ++f) acc[f] += a * w2[e * RD_ + f];
  }
  h16* out = qkfeat + idx * RD_;
#pragma unroll
  for (int f = 0; f < RD_; ++f) out[f] = (h16)acc[f];
}

// W (K x N) f32 -> Wt (N x K) f16
__global__ void transpose_f16_kernel(const float* __restrict__ W, h16* __restrict__ Wt,
                                     int K, int N) {
  int idx = blockIdx.x * blockDim.x + threadIdx.x;
  if (idx >= K * N) return;
  int k = idx / N, n = idx % N;
  Wt[(size_t)n * K + k] = (h16)W[idx];
}

// C(M,N) = act(A(M,K) @ Bt(N,K)^T + bias) ; outputs f32 and/or f16.
// The 8 waves of a block share one mt (requires (N/16) % 8 == 0, true for
// N=256/1024): the 16xK A panel is staged once into LDS via async copies and
// all waves read their A fragments with ds_load broadcasts.
__global__ void gemm_bt_kernel(const h16* __restrict__ A, const h16* __restrict__ Bt,
                               const float* __restrict__ bias,
                               float* __restrict__ Cf, h16* __restrict__ Ch,
                               int M, int N, int K, int relu) {
  __shared__ __align__(16) h16 Apanel[16 * 1024];  // up to K=1024 (32 KB)
  int lane = threadIdx.x;
  int tn = N >> 4;
  int tile0 = blockIdx.x * 8;          // blockDim.y == 8
  int mt = tile0 / tn;                 // uniform across block
  if (mt >= (M >> 4)) return;
  int nt = (tile0 % tn) + threadIdx.y; // tn % 8 == 0 -> in range

  // cooperative async stage of contiguous 16xK A panel
  const h16* Ab = A + (size_t)mt * 16 * K;
  int tid = threadIdx.y * 32 + lane;
  int chunks = (16 * K) / 8;           // 16-byte chunks
  for (int i = tid; i < chunks; i += 256)
    lds_copy16(&Apanel[i * 8], &Ab[i * 8]);
  lds_copy_fence();

  const h16* Bbase = Bt + (size_t)nt * 16 * K;
  v8f c = {};
  for (int kc = 0; kc < K; kc += 32) {
    __builtin_prefetch(Bbase + kc + 64, 0, 1);
    v16h a = ld_a(Apanel + kc, K, lane);   // LDS
    v16h b = ld_bt(Bbase + kc, K, lane);   // global (K-pairs contiguous)
    c = WMMA_F16(a, b, c);
  }
  int n = nt * 16 + (lane & 15), hi = (lane >> 4) & 1;
  float bv = bias ? bias[n] : 0.f;
#pragma unroll
  for (int vv = 0; vv < 8; ++vv) {
    int m = mt * 16 + vv + 8 * hi;
    float x = c[vv] + bv;
    if (relu) x = fmaxf(x, 0.f);
    size_t o = (size_t)m * N + n;
    if (Cf) Cf[o] = x;
    if (Ch) Ch[o] = (h16)x;
  }
}

// scores[b,h,r,s] = (q·k)/sqrt(HD) + qkfeat[b,r,s,:] @ relproj[l][:,h]
__global__ void attn_scores_kernel(const h16* __restrict__ q, const h16* __restrict__ k,
                                   const h16* __restrict__ qkfeat,
                                   const float* __restrict__ relproj_l,
                                   float* __restrict__ scores) {
  const int TN = R_ / 16;  // 24
  int lane = threadIdx.x;
  long tile = (long)blockIdx.x * blockDim.y + threadIdx.y;
  int nt = (int)(tile % TN); tile /= TN;
  int mt = (int)(tile % TN); tile /= TN;
  int hh = (int)(tile % H_);
  int b  = (int)(tile / H_);
  if (b >= B_) return;
  const h16* Aq = q + (size_t)b * R_ * E_ + (size_t)mt * 16 * E_ + hh * HD_;
  const h16* Bk = k + (size_t)b * R_ * E_ + (size_t)nt * 16 * E_ + hh * HD_;
  v16h a = ld_a(Aq, E_, lane);     // K = HD = 32: exactly one WMMA
  v16h bb = ld_bt(Bk, E_, lane);
  v8f c = {};
  c = WMMA_F16(a, bb, c);
  const float scale = 0.17677669529663687f;  // 1/sqrt(32)
  int n = nt * 16 + (lane & 15), hi = (lane >> 4) & 1;
  float pr[RD_];
#pragma unroll
  for (int f = 0; f < RD_; ++f) pr[f] = relproj_l[f * H_ + hh];
#pragma unroll
  for (int vv = 0; vv < 8; ++vv) {
    int m = mt * 16 + vv + 8 * hi;
    const h16* qf = qkfeat + (((size_t)b * R_ + m) * R_ + n) * RD_;
    float bias = 0.f;
#pragma unroll
    for (int f = 0; f < RD_; ++f) bias += (float)qf[f] * pr[f];
    scores[(((size_t)b * H_ + hh) * R_ + m) * R_ + n] = c[vv] * scale + bias;
  }
}

// row-wise softmax over s (R=384 = 12 per lane), write f16 probs
__global__ void softmax_kernel(const float* __restrict__ scores, h16* __restrict__ probs) {
  int lane = threadIdx.x;
  long row = (long)blockIdx.x * blockDim.y + threadIdx.y;
  if (row >= (long)B_ * H_ * R_) return;
  const float* sr = scores + (size_t)row * R_;
  float x[12];
  float mx = -1e30f;
#pragma unroll
  for (int i = 0; i < 12; ++i) { x[i] = sr[i * 32 + lane]; mx = fmaxf(mx, x[i]); }
  mx = wave_max(mx);
  float sum = 0.f;
#pragma unroll
  for (int i = 0; i < 12; ++i) { x[i] = __expf(x[i] - mx); sum += x[i]; }
  sum = wave_sum(sum);
  float inv = 1.f / sum;
  h16* pr = probs + (size_t)row * R_;
#pragma unroll
  for (int i = 0; i < 12; ++i) pr[i * 32 + lane] = (h16)(x[i] * inv);
}

// attn_out[b,r,h*HD+hd] = probs[b,h,r,:] @ v[b,:,h,hd]
__global__ void attn_av_kernel(const h16* __restrict__ probs, const h16* __restrict__ v,
                               h16* __restrict__ attn_out) {
  int lane = threadIdx.x;
  long tile = (long)blockIdx.x * blockDim.y + threadIdx.y;
  int nt = (int)(tile & 1); tile >>= 1;
  int mt = (int)(tile % (R_ / 16)); tile /= (R_ / 16);
  int hh = (int)(tile % H_);
  int b  = (int)(tile / H_);
  if (b >= B_) return;
  const h16* Ap = probs + (((size_t)b * H_ + hh) * R_ + (size_t)mt * 16) * R_;
  const h16* Bv = v + (size_t)b * R_ * E_ + hh * HD_ + nt * 16;
  v8f c = {};
  for (int kc = 0; kc < R_; kc += 32) {
    v16h a = ld_a(Ap + kc, R_, lane);
    v16h bb = ld_b(Bv + (size_t)kc * E_, E_, lane);
    c = WMMA_F16(a, bb, c);
  }
  int n = nt * 16 + (lane & 15), hi = (lane >> 4) & 1;
#pragma unroll
  for (int vv = 0; vv < 8; ++vv) {
    int m = mt * 16 + vv + 8 * hi;
    attn_out[((size_t)b * R_ + m) * E_ + hh * HD_ + n] = (h16)c[vv];
  }
}

// h = LN(h + delta) * g + b ; one wave per row, E=256 -> 8 per lane
__global__ void add_ln_kernel(float* __restrict__ h, h16* __restrict__ hh,
                              const float* __restrict__ delta,
                              const float* __restrict__ g, const float* __restrict__ bta) {
  int lane = threadIdx.x;
  long row = (long)blockIdx.x * blockDim.y + threadIdx.y;
  if (row >= (long)BR_) return;
  size_t base = (size_t)row * E_;
  float x[8];
  float s = 0.f;
#pragma unroll
  for (int i = 0; i < 8; ++i) {
    x[i] = h[base + i * 32 + lane] + delta[base + i * 32 + lane];
    s += x[i];
  }
  s = wave_sum(s);
  float mean = s * (1.f / E_);
  float s2 = 0.f;
#pragma unroll
  for (int i = 0; i < 8; ++i) { float dd = x[i] - mean; s2 += dd * dd; }
  s2 = wave_sum(s2);
  float inv = rsqrtf(s2 * (1.f / E_) + 1e-5f);
#pragma unroll
  for (int i = 0; i < 8; ++i) {
    int col = i * 32 + lane;
    float y = (x[i] - mean) * inv * g[col] + bta[col];
    h[base + col]  = y;
    hh[base + col] = (h16)y;
  }
}

// value head stage 1: per-row relu(h@Wval1+b)@Wval2 + bval2
__global__ void value_row_kernel(const float* __restrict__ h,
                                 const float* __restrict__ Wval1, const float* __restrict__ bval1,
                                 const float* __restrict__ Wval2, const float* __restrict__ bval2,
                                 float* __restrict__ part) {
  __shared__ __align__(16) float hrow[E_];
  __shared__ float red[E_];
  int row = blockIdx.x, t = threadIdx.x;
  if (t < E_ / 4) lds_copy16(&hrow[t * 4], &h[(size_t)row * E_ + t * 4]);
  lds_copy_fence();
  float acc = bval1[t];
  for (int e = 0; e < E_; ++e) acc += hrow[e] * Wval1[(size_t)e * E_ + t];
  acc = fmaxf(acc, 0.f) * Wval2[t];
  red[t] = acc;
  __syncthreads();
  for (int off = E_ / 2; off; off >>= 1) {
    if (t < off) red[t] += red[t + off];
    __syncthreads();
  }
  if (t == 0) part[row] = red[0] + bval2[0];
}

__global__ void value_reduce_kernel(const float* __restrict__ part, float* __restrict__ out) {
  int b = blockIdx.x, lane = threadIdx.x;
  float s = 0.f;
  for (int i = lane; i < R_; i += 32) s += part[(size_t)b * R_ + i];
  s = wave_sum(s);
  if (lane == 0) out[b] = s * (1.f / R_);
}

// nearest station (first-min = argmin semantics)
__global__ void station_kernel(const float* __restrict__ coords,
                               const float* __restrict__ fc, const float* __restrict__ tc,
                               const int* __restrict__ station_idx,
                               float* __restrict__ st1, float* __restrict__ st2) {
  int idx = blockIdx.x * blockDim.x + threadIdx.x;
  if (idx >= BR_) return;
  int b = idx / R_;
  float fx = fc[idx * 2], fy = fc[idx * 2 + 1];
  float tx = tc[idx * 2], ty = tc[idx * 2 + 1];
  float bd1 = 1e30f, bd2 = 1e30f;
  int s1 = 0, s2 = 0;
  for (int j = 0; j < S_; ++j) {
    int node = station_idx[b * (S_ + 1) + j];
    float sx = coords[((size_t)b * NN_ + node) * 2 + 0];
    float sy = coords[((size_t)b * NN_ + node) * 2 + 1];
    float dx1 = fx - sx, dy1 = fy - sy;
    float d1 = sqrtf(dx1 * dx1 + dy1 * dy1);
    if (d1 < bd1) { bd1 = d1; s1 = node; }
    float dx2 = sx - tx, dy2 = sy - ty;
    float d2 = sqrtf(dx2 * dx2 + dy2 * dy2);
    if (d2 < bd2) { bd2 = d2; s2 = node; }
  }
  st1[idx] = (float)s1;
  st2[idx] = (float)s2;
}

// ---------------- Host orchestration ---------------------------------------

extern "C" void kernel_launch(void* const* d_in, const int* in_sizes, int n_in,
                              void* d_out, int out_size, void* d_ws, size_t ws_size,
                              hipStream_t stream) {
  const float* coords = (const float*)d_in[0];
  const float* Wreq   = (const float*)d_in[1];
  const float* breq   = (const float*)d_in[2];
  const float* relW1  = (const float*)d_in[3];
  const float* relb1  = (const float*)d_in[4];
  const float* relW2  = (const float*)d_in[5];
  const float* relb2  = (const float*)d_in[6];
  const float* Wq     = (const float*)d_in[7];
  const float* Wk     = (const float*)d_in[8];
  const float* Wvv    = (const float*)d_in[9];
  const float* Wo     = (const float*)d_in[10];
  const float* relproj= (const float*)d_in[11];
  const float* ln1g   = (const float*)d_in[12];
  const float* ln1b   = (const float*)d_in[13];
  const float* ln2g   = (const float*)d_in[14];
  const float* ln2b   = (const float*)d_in[15];
  const float* Wff1   = (const float*)d_in[16];
  const float* bff1   = (const float*)d_in[17];
  const float* Wff2   = (const float*)d_in[18];
  const float* bff2   = (const float*)d_in[19];
  const float* Wval1  = (const float*)d_in[20];
  const float* bval1  = (const float*)d_in[21];
  const float* Wval2  = (const float*)d_in[22];
  const float* bval2  = (const float*)d_in[23];
  const int* from_idx = (const int*)d_in[24];
  const int* to_idx   = (const int*)d_in[25];
  const int* station_idx = (const int*)d_in[26];

  // workspace layout
  char* w = (char*)d_ws;
  size_t off = 0;
  auto alloc = [&](size_t bytes) -> char* {
    char* p = w + off;
    off = (off + bytes + 255) & ~(size_t)255;
    return p;
  };
  float* h      = (float*)alloc((size_t)BR_ * E_ * 4);
  h16*   hh     = (h16*)  alloc((size_t)BR_ * E_ * 2);
  h16*   qh     = (h16*)  alloc((size_t)BR_ * E_ * 2);
  h16*   kh     = (h16*)  alloc((size_t)BR_ * E_ * 2);
  h16*   vh     = (h16*)  alloc((size_t)BR_ * E_ * 2);
  h16*   ao     = (h16*)  alloc((size_t)BR_ * E_ * 2);
  float* delta  = (float*)alloc((size_t)BR_ * E_ * 4);
  h16*   ff1    = (h16*)  alloc((size_t)BR_ * 4 * E_ * 2);
  float* scores = (float*)alloc((size_t)B_ * H_ * R_ * R_ * 4);
  h16*   probs  = (h16*)  alloc((size_t)B_ * H_ * R_ * R_ * 2);
  h16*   qkfeat = (h16*)  alloc((size_t)B_ * R_ * R_ * RD_ * 2);
  float* fcb    = (float*)alloc((size_t)BR_ * 2 * 4);
  float* tcb    = (float*)alloc((size_t)BR_ * 2 * 4);
  float* vpart  = (float*)alloc((size_t)BR_ * 4);
  h16*   wtbase = (h16*)  alloc((size_t)L_ * (4 * E_ * E_ + 8 * E_ * E_) * 2);

  h16 *WqT[L_], *WkT[L_], *WvT[L_], *WoT[L_], *Wff1T[L_], *Wff2T[L_];
  {
    h16* p = wtbase;
    for (int l = 0; l < L_; ++l) {
      WqT[l] = p;   p += (size_t)E_ * E_;
      WkT[l] = p;   p += (size_t)E_ * E_;
      WvT[l] = p;   p += (size_t)E_ * E_;
      WoT[l] = p;   p += (size_t)E_ * E_;
      Wff1T[l] = p; p += (size_t)E_ * 4 * E_;
      Wff2T[l] = p; p += (size_t)4 * E_ * E_;
    }
  }

  float* out_h   = (float*)d_out;
  float* out_val = out_h + (size_t)BR_ * E_;
  float* out_s1  = out_val + B_;
  float* out_s2  = out_s1 + BR_;

  dim3 wblk(32, 8);

  // 1. embed
  embed_kernel<<<(BR_ * E_ + 255) / 256, 256, 0, stream>>>(
      coords, from_idx, to_idx, Wreq, breq, h, hh, fcb, tcb);

  // 2. qkfeat (computed once, reused by all layers)
  qkfeat_kernel<<<((size_t)B_ * R_ * R_ + 255) / 256, 256, 0, stream>>>(
      fcb, relW1, relb1, relW2, relb2, qkfeat);

  // 3. convert + transpose all layer weights to f16 (N x K)
  for (int l = 0; l < L_; ++l) {
    int ee = E_ * E_;
    transpose_f16_kernel<<<(ee + 255) / 256, 256, 0, stream>>>(Wq  + (size_t)l * ee, WqT[l], E_, E_);
    transpose_f16_kernel<<<(ee + 255) / 256, 256, 0, stream>>>(Wk  + (size_t)l * ee, WkT[l], E_, E_);
    transpose_f16_kernel<<<(ee + 255) / 256, 256, 0, stream>>>(Wvv + (size_t)l * ee, WvT[l], E_, E_);
    transpose_f16_kernel<<<(ee + 255) / 256, 256, 0, stream>>>(Wo  + (size_t)l * ee, WoT[l], E_, E_);
    int ef = E_ * 4 * E_;
    transpose_f16_kernel<<<(ef + 255) / 256, 256, 0, stream>>>(Wff1 + (size_t)l * ef, Wff1T[l], E_, 4 * E_);
    transpose_f16_kernel<<<(ef + 255) / 256, 256, 0, stream>>>(Wff2 + (size_t)l * ef, Wff2T[l], 4 * E_, E_);
  }

  auto gemm = [&](const h16* A, const h16* Bt, const float* bias,
                  float* Cf, h16* Ch, int M, int N, int K, int relu) {
    int tiles = (M / 16) * (N / 16);
    gemm_bt_kernel<<<(tiles + 7) / 8, wblk, 0, stream>>>(A, Bt, bias, Cf, Ch, M, N, K, relu);
  };

  // 4. transformer layers
  for (int l = 0; l < L_; ++l) {
    gemm(hh, WqT[l], nullptr, nullptr, qh, BR_, E_, E_, 0);
    gemm(hh, WkT[l], nullptr, nullptr, kh, BR_, E_, E_, 0);
    gemm(hh, WvT[l], nullptr, nullptr, vh, BR_, E_, E_, 0);

    {
      long tiles = (long)B_ * H_ * (R_ / 16) * (R_ / 16);
      attn_scores_kernel<<<(unsigned)((tiles + 7) / 8), wblk, 0, stream>>>(
          qh, kh, qkfeat, relproj + (size_t)l * RD_ * H_, scores);
    }
    {
      long rows = (long)B_ * H_ * R_;
      softmax_kernel<<<(unsigned)((rows + 7) / 8), wblk, 0, stream>>>(scores, probs);
    }
    {
      long tiles = (long)B_ * H_ * (R_ / 16) * 2;
      attn_av_kernel<<<(unsigned)((tiles + 7) / 8), wblk, 0, stream>>>(probs, vh, ao);
    }
    gemm(ao, WoT[l], nullptr, delta, nullptr, BR_, E_, E_, 0);
    add_ln_kernel<<<(BR_ + 7) / 8, wblk, 0, stream>>>(
        h, hh, delta, ln1g + (size_t)l * E_, ln1b + (size_t)l * E_);

    gemm(hh, Wff1T[l], bff1 + (size_t)l * 4 * E_, nullptr, ff1, BR_, 4 * E_, E_, 1);
    gemm(ff1, Wff2T[l], bff2 + (size_t)l * E_, delta, nullptr, BR_, E_, 4 * E_, 0);
    add_ln_kernel<<<(BR_ + 7) / 8, wblk, 0, stream>>>(
        h, hh, delta, ln2g + (size_t)l * E_, ln2b + (size_t)l * E_);
  }

  // 5. value head
  value_row_kernel<<<BR_, E_, 0, stream>>>(h, Wval1, bval1, Wval2, bval2, vpart);
  value_reduce_kernel<<<B_, 32, 0, stream>>>(vpart, out_val);

  // 6. stations
  station_kernel<<<(BR_ + 255) / 256, 256, 0, stream>>>(
      coords, fcb, tcb, station_idx, out_s1, out_s2);

  // 7. final h -> d_out
  (void)hipMemcpyAsync(out_h, h, (size_t)BR_ * E_ * sizeof(float),
                       hipMemcpyDeviceToDevice, stream);
}
